// LSTM_Sequence_Prediction_43834436223776
// MI455X (gfx1250) — matrix operand order repaired and checked by reading, not compiled, not run on hardware
//
#include <hip/hip_runtime.h>

// ---------------------------------------------------------------------------
// LSTM sequence prediction for MI455X (gfx1250, wave32, WMMA + TDM).
// Encoder: z = GN(concat(x_flat, h0) @ enc_W^T + enc_b); h,c = gates(z, c0)
// Decoder: T x { z = GN(h @ dec_W^T + dec_b); h,c = gates; out = h @ out_W^T + out_b }
// GEMMs on v_wmma_f32_16x16x32_bf16 (bf16 in, f32 accumulate).
// Decoder weights staged into LDS per block via tensor_load_to_lds (TDM).
// ---------------------------------------------------------------------------

typedef __bf16 bf16;
typedef __attribute__((ext_vector_type(16))) __bf16 v16bf;
typedef __attribute__((ext_vector_type(8)))  __bf16 v8bf;
typedef __attribute__((ext_vector_type(8)))  float  v8f;
typedef __attribute__((ext_vector_type(4)))  unsigned v4u;
typedef __attribute__((ext_vector_type(8)))  unsigned v8u;

#define B_DIM   128
#define H_DIM   1024
#define FOUR_H  4096
#define D_DIM   256
#define S_DIM   64
#define K_X     (S_DIM * D_DIM)     // 16384
#define K_ENC   (K_X + H_DIM)       // 17408

// ---- WMMA fragment loaders -------------------------------------------------
// A matrix (16x32, bf16): lanes 0-15 -> rows 0-15, one contiguous 16-element K run
// at 16*(lane>>4).  B matrix (32x16): lane = column N, one contiguous
// 16-element K run at 16*(lane>>4).  Sources are row-major [rows, K].

__device__ __forceinline__ v16bf load_a_bf(const bf16* __restrict__ base, int ld, int k0) {
    const int lane = threadIdx.x & 31;
    const int r    = lane & 15;
    const int kb   = (lane >> 4) << 4;
    const bf16* p  = base + r * ld + k0 + kb;
    v8bf lo = *(const v8bf*)(p);
    v8bf hi = *(const v8bf*)(p + 8);
    return __builtin_shufflevector(lo, hi, 0,1,2,3,4,5,6,7,8,9,10,11,12,13,14,15);
}

// Same B-fragment gather but from an LDS-resident tile (emits ds_load).
__device__ __forceinline__ v16bf load_b_lds(const bf16* base, int ld, int k0) {
    const int lane = threadIdx.x & 31;
    const int n    = lane & 15;
    const int kb   = (lane >> 4) << 4;
    return *(const v16bf*)(base + n * ld + k0 + kb);
}

__device__ __forceinline__ v16bf load_b_bf(const bf16* __restrict__ base, int ld, int k0) {
    const int lane = threadIdx.x & 31;
    const int n    = lane & 15;
    const int kb   = (lane >> 4) << 4;
    return *(const v16bf*)(base + n * ld + k0 + kb);
}

__device__ __forceinline__ v16bf load_a_f32(const float* __restrict__ base, int ld, int k0) {
    const int lane = threadIdx.x & 31;
    const int r    = lane & 15;
    const int kb   = (lane >> 4) << 4;
    const float* p = base + r * ld + k0 + kb;
    v16bf out;
#pragma unroll
    for (int i = 0; i < 8; ++i) out[i]     = (bf16)p[i];
#pragma unroll
    for (int i = 0; i < 8; ++i) out[8 + i] = (bf16)p[8 + i];
    return out;
}

__device__ __forceinline__ v16bf load_b_f32(const float* __restrict__ base, int ld, int k0) {
    const int lane = threadIdx.x & 31;
    const int n    = lane & 15;
    const int kb   = (lane >> 4) << 4;
    const float* p = base + n * ld + k0 + kb;
    v16bf out;
#pragma unroll
    for (int i = 0; i < 16; ++i) out[i] = (bf16)p[i];
    return out;
}

__device__ __forceinline__ v8f wmma_bf(v16bf a, v16bf b, v8f c) {
    return __builtin_amdgcn_wmma_f32_16x16x32_bf16(
        /*neg_a=*/false, a, /*neg_b=*/false, b,
        /*c_mod=*/(short)0, c, /*reuse_a=*/false, /*reuse_b=*/false);
}

// f32 C/D 16x16: VGPR j -> row (8*(lane>>4) + j), column = lane&15.
__device__ __forceinline__ void store_tile(float* __restrict__ dst, int ldd,
                                           v8f acc, const float* __restrict__ bias_n0) {
    const int lane = threadIdx.x & 31;
    const int n    = lane & 15;
    const int mo   = (lane >> 4) << 3;
    const float bv = bias_n0[n];
#pragma unroll
    for (int j = 0; j < 8; ++j) dst[(mo + j) * ldd + n] = acc[j] + bv;
}

// ---- TDM: 1-D tensor DMA of a contiguous global chunk into LDS -------------
// D# built per CDNA5 ISA ch.8: group0 = {count=1, lds_addr, global_addr, type=2},
// group1 = {data_size=8B, tensor_dim0 = tile_dim0 = nbytes/8, dim1 = 1}.
// Issued by one wave; completion via s_wait_tensorcnt.
__device__ __forceinline__ void tdm_load_1d(const void* lds_ptr, const void* gptr,
                                            unsigned nbytes) {
    const unsigned units   = nbytes >> 3;                        // 8-byte units
    const unsigned lds_off = (unsigned)(unsigned long long)lds_ptr;
    const unsigned long long ga = (unsigned long long)gptr;
    v4u g0;
    g0[0] = 1u;                                                  // count=1 (valid user D#)
    g0[1] = lds_off;                                             // lds_addr (bytes)
    g0[2] = (unsigned)(ga & 0xFFFFFFFFu);                        // global_addr[31:0]
    g0[3] = (unsigned)((ga >> 32) & 0x01FFFFFFu) | (2u << 30);   // addr[56:32] | type=2
    v8u g1;
    g1[0] = (3u << 16);                    // workgroup_mask=0, data_size=3 (8B)
    g1[1] = (units & 0xFFFFu) << 16;       // tensor_dim0[15:0]
    g1[2] = (units >> 16) | (1u << 16);    // tensor_dim0[31:16] | tensor_dim1 lo = 1
    g1[3] = (units & 0xFFFFu) << 16;       // tensor_dim1 hi = 0 | tile_dim0 = units
    g1[4] = 1u;                            // tile_dim1 = 1, tile_dim2 = 0
    g1[5] = units;                         // tensor_dim0_stride[31:0]
    g1[6] = 0u;                            // stride hi, tensor_dim1_stride lo
    g1[7] = 0u;
    asm volatile("tensor_load_to_lds %0, %1" :: "s"(g0), "s"(g1) : "memory");
}

// ---- fp32 -> bf16 convert (dec_W / out_W, done once, stays in L2) ----------
__global__ void cvt_f32_bf16_kernel(const float* __restrict__ src,
                                    bf16* __restrict__ dst, int n) {
    int i = blockIdx.x * blockDim.x + threadIdx.x;
    const int stride = gridDim.x * blockDim.x;
    for (; i < n; i += stride) dst[i] = (bf16)src[i];
}

// ---- Encoder GEMM: z = concat(x_flat, h0) @ enc_W^T + enc_b ----------------
// grid.x = 4096/16 = 256 (N tiles), block = 256 threads = 8 waves (M tiles).
// HBM-bound on the 285 MB enc_W stream; bf16 conversion happens in registers.
__global__ void enc_gemm_kernel(const float* __restrict__ x,
                                const float* __restrict__ h0,
                                const float* __restrict__ W,
                                const float* __restrict__ bias,
                                float* __restrict__ z) {
    const int m0 = (threadIdx.x >> 5) << 4;   // wave id * 16
    const int n0 = blockIdx.x << 4;
    const float* Arow = x  + m0 * K_X;        // x viewed as [128, 16384]
    const float* Hrow = h0 + m0 * H_DIM;
    const float* Wrow = W  + n0 * K_ENC;
    const int lane = threadIdx.x & 31;

    v8f acc = {};
    for (int k0 = 0; k0 < K_X; k0 += 32) {
        // keep the weight stream ahead of us (global_prefetch_b8)
        __builtin_prefetch((const void*)(Wrow + (lane & 15) * K_ENC + k0 + 512), 0, 1);
        v16bf a = load_a_f32(Arow, K_X, k0);
        v16bf b = load_b_f32(Wrow, K_ENC, k0);
        acc = wmma_bf(a, b, acc);
    }
#pragma unroll 4
    for (int k0 = 0; k0 < H_DIM; k0 += 32) {
        v16bf a = load_a_f32(Hrow, H_DIM, k0);
        v16bf b = load_b_f32(Wrow, K_ENC, K_X + k0);
        acc = wmma_bf(a, b, acc);
    }
    store_tile(z + m0 * FOUR_H + n0, FOUR_H, acc, bias + n0);
}

// ---- Decoder GEMM with TDM-staged weights ----------------------------------
// z[128,4096] = h @ dec_W^T + dec_b.  grid.x = 4096/32 = 128 blocks.
// Each block TDM-copies its 32 N-rows of dec_W (64 KB contiguous, bf16) into
// LDS once, then runs the whole K=1024 loop with B-fragments from LDS.
// Waves: 2 (N) x 4 (M-groups), 2 M-tiles each -> 2 WMMA per 3 fragment loads.
__global__ void dec_gemm_tdm_kernel(const bf16* __restrict__ A,     // h  [128,1024]
                                    const bf16* __restrict__ W,     // [4096,1024]
                                    const float* __restrict__ bias,
                                    float* __restrict__ dst) {      // z  [128,4096]
    __shared__ bf16 sW[32 * H_DIM];                                 // 64 KB
    const int n_base = blockIdx.x << 5;

    if ((threadIdx.x >> 5) == 0) {                                  // one wave issues DMA
        tdm_load_1d(&sW[0], W + (size_t)n_base * H_DIM,
                    32u * H_DIM * sizeof(bf16));
        __builtin_amdgcn_s_wait_tensorcnt(0);
    }
    __syncthreads();

    const int wave    = threadIdx.x >> 5;
    const int n_wave  = wave & 1;            // 2 N-waves
    const int m_group = wave >> 1;           // 4 M-groups
    const int m0      = m_group << 5;        // 2 M-tiles per wave
    const int n0loc   = n_wave << 4;

    const bf16* A0 = A + m0 * H_DIM;
    const bf16* A1 = A + (m0 + 16) * H_DIM;
    const bf16* Wl = &sW[n0loc * H_DIM];

    v8f acc0 = {}, acc1 = {};
#pragma unroll 2
    for (int k0 = 0; k0 < H_DIM; k0 += 32) {
        v16bf b = load_b_lds(Wl, H_DIM, k0);                  // ds_load from LDS
        acc0 = wmma_bf(load_a_bf(A0, H_DIM, k0), b, acc0);
        acc1 = wmma_bf(load_a_bf(A1, H_DIM, k0), b, acc1);
    }
    const int n0 = n_base + n0loc;
    store_tile(dst + m0 * FOUR_H + n0,        FOUR_H, acc0, bias + n0);
    store_tile(dst + (m0 + 16) * FOUR_H + n0, FOUR_H, acc1, bias + n0);
}

// ---- Generic bf16 NT GEMM + bias (used for the small output projection) ----
__global__ void gemm_bf16_kernel(const bf16* __restrict__ A, int lda,
                                 const bf16* __restrict__ W, int ldw,
                                 const float* __restrict__ bias,
                                 float* __restrict__ dst, int ldd, int K) {
    const int m0 = (threadIdx.x >> 5) << 4;
    const int n0 = blockIdx.x << 4;
    const bf16* Ab = A + m0 * lda;
    const bf16* Wb = W + n0 * ldw;
    v8f acc = {};
#pragma unroll 4
    for (int k0 = 0; k0 < K; k0 += 32) {
        acc = wmma_bf(load_a_bf(Ab, lda, k0), load_b_bf(Wb, ldw, k0), acc);
    }
    store_tile(dst + m0 * ldd + n0, ldd, acc, bias + n0);
}

// ---- Fused GroupNorm(4 groups == i/f/o/g gates) + LSTM gates ---------------
__device__ __forceinline__ float sigm_f(float x) { return 1.f / (1.f + __expf(-x)); }
__device__ __forceinline__ float tanh_f(float x) {
    float e = __expf(2.f * x);
    return (e - 1.f) / (e + 1.f);
}

__global__ void norm_gates_kernel(const float* __restrict__ z,
                                  const float* __restrict__ gamma,
                                  const float* __restrict__ beta,
                                  const float* __restrict__ c_in,
                                  float* __restrict__ c_out,
                                  bf16* __restrict__ h_bf) {
    const int b   = blockIdx.x;
    const int tid = threadIdx.x;             // 0..255
    const float* zb = z + b * FOUR_H;

    float v[4][4];
    float psum[4], psq[4];
#pragma unroll
    for (int g = 0; g < 4; ++g) {
        psum[g] = 0.f; psq[g] = 0.f;
#pragma unroll
        for (int r = 0; r < 4; ++r) {
            float t = zb[g * H_DIM + r * 256 + tid];
            v[g][r] = t;
            psum[g] += t;
            psq[g]  += t * t;
        }
    }

    __shared__ float sred[8][256];
#pragma unroll
    for (int g = 0; g < 4; ++g) { sred[g][tid] = psum[g]; sred[4 + g][tid] = psq[g]; }
    __syncthreads();
    for (int off = 128; off > 0; off >>= 1) {
        if (tid < off) {
#pragma unroll
            for (int k = 0; k < 8; ++k) sred[k][tid] += sred[k][tid + off];
        }
        __syncthreads();
    }

    float mu[4], rs[4];
#pragma unroll
    for (int g = 0; g < 4; ++g) {
        mu[g] = sred[g][0] * (1.f / 1024.f);
        float var = sred[4 + g][0] * (1.f / 1024.f) - mu[g] * mu[g];
        rs[g] = rsqrtf(var + 1e-5f);
    }

#pragma unroll
    for (int r = 0; r < 4; ++r) {
        const int j = r * 256 + tid;
        float zi = (v[0][r] - mu[0]) * rs[0] * gamma[j]              + beta[j];
        float zf = (v[1][r] - mu[1]) * rs[1] * gamma[H_DIM + j]      + beta[H_DIM + j];
        float zo = (v[2][r] - mu[2]) * rs[2] * gamma[2 * H_DIM + j]  + beta[2 * H_DIM + j];
        float zg = (v[3][r] - mu[3]) * rs[3] * gamma[3 * H_DIM + j]  + beta[3 * H_DIM + j];
        float cprev = c_in[b * H_DIM + j];
        float cn = sigm_f(zf) * cprev + sigm_f(zi) * tanh_f(zg);
        c_out[b * H_DIM + j] = cn;
        h_bf[b * H_DIM + j]  = (bf16)(sigm_f(zo) * tanh_f(cn));
    }
}

// ---------------------------------------------------------------------------
extern "C" void kernel_launch(void* const* d_in, const int* in_sizes, int n_in,
                              void* d_out, int out_size, void* d_ws, size_t ws_size,
                              hipStream_t stream) {
    const float* x       = (const float*)d_in[0];   // [64,128,256]
    const float* h0      = (const float*)d_in[1];   // [128,1024]
    const float* c0      = (const float*)d_in[2];   // [128,1024]
    const float* encW    = (const float*)d_in[3];   // [4096,17408]
    const float* encb    = (const float*)d_in[4];
    const float* encg    = (const float*)d_in[5];
    const float* encbeta = (const float*)d_in[6];
    const float* decW    = (const float*)d_in[7];   // [4096,1024]
    const float* decb    = (const float*)d_in[8];
    const float* decg    = (const float*)d_in[9];
    const float* decbeta = (const float*)d_in[10];
    const float* outW    = (const float*)d_in[11];  // [256,1024]
    const float* outb    = (const float*)d_in[12];
    (void)in_sizes; (void)n_in; (void)ws_size;

    float* out = (float*)d_out;
    const int T = out_size / (B_DIM * D_DIM);       // target_len (host-side, capture-safe)

    // Workspace carve-up (all regions pow2-aligned; ~12 MB total)
    char* ws = (char*)d_ws;
    float* z      = (float*)ws;  ws += (size_t)B_DIM * FOUR_H * sizeof(float);   // 2 MB
    float* c      = (float*)ws;  ws += (size_t)B_DIM * H_DIM * sizeof(float);    // 512 KB
    bf16*  hbf    = (bf16*)ws;   ws += (size_t)B_DIM * H_DIM * sizeof(bf16);     // 256 KB
    bf16*  decWbf = (bf16*)ws;   ws += (size_t)FOUR_H * H_DIM * sizeof(bf16);    // 8 MB
    bf16*  outWbf = (bf16*)ws;   ws += (size_t)D_DIM * H_DIM * sizeof(bf16);     // 512 KB

    // One-time bf16 weight copies for the reused decoder weights (live in L2).
    cvt_f32_bf16_kernel<<<2048, 256, 0, stream>>>(decW, decWbf, FOUR_H * H_DIM);
    cvt_f32_bf16_kernel<<<128,  256, 0, stream>>>(outW, outWbf, D_DIM * H_DIM);

    // Encoder: HBM-bound 285 MB weight stream, bf16-convert in registers.
    enc_gemm_kernel<<<FOUR_H / 16, 256, 0, stream>>>(x, h0, encW, encb, z);
    norm_gates_kernel<<<B_DIM, 256, 0, stream>>>(z, encg, encbeta, c0, c, hbf);

    // Decoder: T sequential steps; dec_W staged per-block into LDS via TDM.
    for (int t = 0; t < T; ++t) {
        dec_gemm_tdm_kernel<<<FOUR_H / 32, 256, 0, stream>>>(hbf, decWbf, decb, z);
        norm_gates_kernel<<<B_DIM, 256, 0, stream>>>(z, decg, decbeta, c, c, hbf);
        gemm_bf16_kernel<<<D_DIM / 16, 256, 0, stream>>>(hbf, H_DIM, outWbf, H_DIM,
                                                         outb, out + (size_t)t * B_DIM * D_DIM,
                                                         D_DIM, H_DIM);
    }
}
